// Block_8916352107084
// MI455X (gfx1250) — matrix operand order repaired and checked by reading, not compiled, not run on hardware
//
#include <hip/hip_runtime.h>
#include <hip/hip_bf16.h>

typedef __attribute__((ext_vector_type(8)))  _Float16 v8h;
typedef __attribute__((ext_vector_type(16))) _Float16 v16h;
typedef __attribute__((ext_vector_type(8)))  float    v8f;

#define WMMA_F32_F16(a, b, c) \
    __builtin_amdgcn_wmma_f32_16x16x32_f16(false, (a), false, (b), (short)0, (c), false, false)

// ---------------------------------------------------------------------------
// 16-lane butterfly reductions (within each half of a wave32) via
// v_permlane16_b32 (pure VALU, no LDS round-trip).
// ---------------------------------------------------------------------------
#if __has_builtin(__builtin_amdgcn_permlane16)
#define PERMXF(v, LO, HI)                                                     \
    __builtin_bit_cast(float, __builtin_amdgcn_permlane16(                    \
        __builtin_bit_cast(int, (v)), __builtin_bit_cast(int, (v)),           \
        (int)(LO), (int)(HI), false, false))
__device__ __forceinline__ float redmax16(float v) {
    v = fmaxf(v, PERMXF(v, 0x67452301u, 0xEFCDAB89u));  // xor 1
    v = fmaxf(v, PERMXF(v, 0x54761032u, 0xDCFE98BAu));  // xor 2
    v = fmaxf(v, PERMXF(v, 0x32107654u, 0xBA98FEDCu));  // xor 4
    v = fmaxf(v, PERMXF(v, 0xFEDCBA98u, 0x76543210u));  // xor 8
    return v;
}
__device__ __forceinline__ float redsum16(float v) {
    v += PERMXF(v, 0x67452301u, 0xEFCDAB89u);
    v += PERMXF(v, 0x54761032u, 0xDCFE98BAu);
    v += PERMXF(v, 0x32107654u, 0xBA98FEDCu);
    v += PERMXF(v, 0xFEDCBA98u, 0x76543210u);
    return v;
}
#else
__device__ __forceinline__ float redmax16(float v) {
    for (int m = 1; m < 16; m <<= 1) v = fmaxf(v, __shfl_xor(v, m));
    return v;
}
__device__ __forceinline__ float redsum16(float v) {
    for (int m = 1; m < 16; m <<= 1) v += __shfl_xor(v, m);
    return v;
}
#endif

// ---------------------------------------------------------------------------
// CDNA5 async copy: global -> LDS, 16 bytes per lane (ASYNCcnt-tracked).
// GVS addressing: saddr(base) + vaddr(byte offset). LDS dest address is the
// low 32 bits of the generic shared pointer.
// ---------------------------------------------------------------------------
__device__ __forceinline__ void async_ld16(unsigned lds_addr, unsigned goff_bytes,
                                           const void* base) {
    asm volatile("global_load_async_to_lds_b128 %0, %1, %2"
                 :: "v"(lds_addr), "v"(goff_bytes), "s"(base)
                 : "memory");
}
__device__ __forceinline__ void wait_async0() {
    asm volatile("s_wait_asynccnt 0" ::: "memory");
}

// ---------------------------------------------------------------------------
// Convert f32 weight [K][N] -> f16 transposed [N][K] (column-major for B-frags)
// ---------------------------------------------------------------------------
__global__ void cvt_transpose_f16(const float* __restrict__ w,
                                  _Float16* __restrict__ wt, int K, int N) {
    int idx = blockIdx.x * blockDim.x + threadIdx.x;
    if (idx >= K * N) return;
    int k = idx / N;
    int n = idx - k * N;
    wt[(size_t)n * K + k] = (_Float16)w[idx];
}

// ---------------------------------------------------------------------------
// RMSNorm over C=1024, output f16. One block (256 thr) per row, 4 elems/thread.
// ---------------------------------------------------------------------------
__global__ void rmsnorm_f16(const float* __restrict__ x,
                            const float* __restrict__ gw,
                            _Float16* __restrict__ out) {
    const int row = blockIdx.x;
    const int t = threadIdx.x;
    const float4 xv = ((const float4*)(x + (size_t)row * 1024))[t];
    float ss = xv.x * xv.x + xv.y * xv.y + xv.z * xv.z + xv.w * xv.w;
    ss = redsum16(ss);
    ss += __shfl_xor(ss, 16);
    __shared__ float red[8];
    if ((t & 31) == 0) red[t >> 5] = ss;
    __syncthreads();
    float tot = 0.f;
#pragma unroll
    for (int i = 0; i < 8; ++i) tot += red[i];
    const float rinv = rsqrtf(tot * (1.0f / 1024.0f) + 1e-6f);
    const float4 gv = ((const float4*)gw)[t];
    _Float16* o = out + (size_t)row * 1024 + t * 4;
    o[0] = (_Float16)(xv.x * gv.x * rinv);
    o[1] = (_Float16)(xv.y * gv.y * rinv);
    o[2] = (_Float16)(xv.z * gv.z * rinv);
    o[3] = (_Float16)(xv.w * gv.w * rinv);
}

// ---------------------------------------------------------------------------
// GEMM: out[M][N] = A[M][K](f16) * Bt[N][K](f16, pre-transposed) (+bias)(relu)(+res)
// Block = 256 thr = 8 waves; wave tile 16(M) x 64(N); block tile 64x128.
// K-loop double-buffered through LDS via global_load_async_to_lds_b128:
// per 32-K step the block stages A tile 64x32 (1 B128/thread) and B tile
// 128x32 (2 B128/thread); waves read WMMA fragments from LDS and issue
// 4 v_wmma per step (A-fragment reused 4x).
//   A-frag: lane m=lane%16, g=lane/16: elems 0..7 -> K=g*8+i, 8..15 -> K=16+g*8+i
//   B-frag: lane n=lane%16, g:         elems 0..7 -> K=g*8+i, 8..15 -> K=16+g*8+i
//   C-frag: VGPR r: row = r + 8*g, col = lane%16
// ---------------------------------------------------------------------------
template <bool OF16, bool RELU, bool BIAS, bool RES>
__global__ void gemm_wmma(const _Float16* __restrict__ A,
                          const _Float16* __restrict__ Bt,
                          const float* __restrict__ bias,
                          const float* __restrict__ res,
                          void* __restrict__ outv,
                          int Mdim, int Ndim, int Kdim) {
    (void)Mdim;
    const int t     = threadIdx.x;
    const int lane  = t & 31;
    const int wave  = t >> 5;
    const int lanen = lane & 15;
    const int g     = lane >> 4;
    const int rowBase = blockIdx.y * 64;
    const int colBase = blockIdx.x * 128;

    __shared__ __align__(64) _Float16 As[2][64 * 32];
    __shared__ __align__(64) _Float16 Bs[2][128 * 32];

    // Per-thread async-staging assignment (16B = 8-half chunks).
    const int arow = t >> 2, aoff = (t & 3) * 8;             // A: 1 chunk/thread
    const int bc0 = t, bc1 = t + 256;                        // B: 2 chunks/thread
    const unsigned aLds[2] = {(unsigned)(uintptr_t)&As[0][arow * 32 + aoff],
                              (unsigned)(uintptr_t)&As[1][arow * 32 + aoff]};
    const unsigned bLds0[2] = {(unsigned)(uintptr_t)&Bs[0][(bc0 >> 2) * 32 + (bc0 & 3) * 8],
                               (unsigned)(uintptr_t)&Bs[1][(bc0 >> 2) * 32 + (bc0 & 3) * 8]};
    const unsigned bLds1[2] = {(unsigned)(uintptr_t)&Bs[0][(bc1 >> 2) * 32 + (bc1 & 3) * 8],
                               (unsigned)(uintptr_t)&Bs[1][(bc1 >> 2) * 32 + (bc1 & 3) * 8]};
    const unsigned aGo  = ((unsigned)(rowBase + arow) * (unsigned)Kdim + (unsigned)aoff) * 2u;
    const unsigned bGo0 = ((unsigned)(colBase + (bc0 >> 2)) * (unsigned)Kdim + (unsigned)((bc0 & 3) * 8)) * 2u;
    const unsigned bGo1 = ((unsigned)(colBase + (bc1 >> 2)) * (unsigned)Kdim + (unsigned)((bc1 & 3) * 8)) * 2u;

    // Wave fragment coordinates (LDS-local).
    const int frow = (wave & 3) * 16 + lanen;       // A row within tile
    const int fcolBase = (wave >> 2) * 64 + lanen;  // B col group within tile

    v8f cacc[4] = {{}, {}, {}, {}};
    const int nk = Kdim >> 5;

    // prologue: stage k-step 0 into buffer 0
    async_ld16(aLds[0], aGo, A);
    async_ld16(bLds0[0], bGo0, Bt);
    async_ld16(bLds1[0], bGo1, Bt);

    for (int i = 0; i < nk; ++i) {
        const int cur = i & 1;
        wait_async0();
        __syncthreads();                       // staged tile visible to all waves
        if (i + 1 < nk) {                      // prefetch next k-step
            const unsigned koff = (unsigned)(i + 1) * 64u;  // 32 halves * 2B
            async_ld16(aLds[cur ^ 1], aGo + koff, A);
            async_ld16(bLds0[cur ^ 1], bGo0 + koff, Bt);
            async_ld16(bLds1[cur ^ 1], bGo1 + koff, Bt);
        }
        const _Float16* Ac = As[cur];
        const _Float16* Bc = Bs[cur];
        v8h lo = *(const v8h*)(Ac + frow * 32 + g * 8);
        v8h hi = *(const v8h*)(Ac + frow * 32 + 16 + g * 8);
        v16h a;
#pragma unroll
        for (int j = 0; j < 8; ++j) { a[j] = lo[j]; a[8 + j] = hi[j]; }
#pragma unroll
        for (int nt = 0; nt < 4; ++nt) {
            v8h blo = *(const v8h*)(Bc + (fcolBase + nt * 16) * 32 + g * 8);
            v8h bhi = *(const v8h*)(Bc + (fcolBase + nt * 16) * 32 + 16 + g * 8);
            v16h b;
#pragma unroll
            for (int j = 0; j < 8; ++j) { b[j] = blo[j]; b[8 + j] = bhi[j]; }
            cacc[nt] = WMMA_F32_F16(a, b, cacc[nt]);
        }
        __syncthreads();                       // readers done before buf reuse
    }

#pragma unroll
    for (int nt = 0; nt < 4; ++nt) {
        const int col = colBase + (wave >> 2) * 64 + nt * 16 + lanen;
#pragma unroll
        for (int r = 0; r < 8; ++r) {
            const int row = rowBase + (wave & 3) * 16 + r + 8 * g;
            float v = cacc[nt][r];
            if constexpr (BIAS) v += bias[col];
            if constexpr (RELU) v = fmaxf(v, 0.f);
            if constexpr (RES)  v += res[(size_t)row * Ndim + col];
            if constexpr (OF16) ((_Float16*)outv)[(size_t)row * Ndim + col] = (_Float16)v;
            else                ((float*)outv)[(size_t)row * Ndim + col] = v;
        }
    }
}

// ---------------------------------------------------------------------------
// Flash-style causal attention. Q,K,V,O stored as (B*T, C) f16 with head slice
// at column h*64. One wave = 16 query rows; block = 8 waves = 128 rows.
// Streams 32-key blocks: S = Q K^T via WMMA, fp32 online softmax (permlane16
// reductions), P re-laid-out via per-wave LDS tile, O += P V via WMMA.
// Per-wave trip counts differ (causal) -> all sync is intra-wave.
// ---------------------------------------------------------------------------
__global__ void attn_wmma(const _Float16* __restrict__ Qf,
                          const _Float16* __restrict__ Kf,
                          const _Float16* __restrict__ Vf,
                          _Float16* __restrict__ Of) {
    constexpr int T = 1024, Cc = 1024;
    const int lane  = threadIdx.x & 31;
    const int wave  = threadIdx.x >> 5;
    const int lanen = lane & 15;
    const int g     = lane >> 4;
    const int qbase = blockIdx.x * 128 + wave * 16;
    const int h64   = blockIdx.y * 64;
    const size_t bT = (size_t)blockIdx.z * T;

    __shared__ __align__(64) _Float16 Pbuf[8][16 * 32];
    _Float16* P = Pbuf[wave];

    // Q fragments for DH=64 (two K=32 steps)
    const _Float16* Qr = Qf + (bT + qbase + lanen) * Cc + h64;
    v16h aQ[2];
#pragma unroll
    for (int kk = 0; kk < 2; ++kk) {
        v8h lo = *(const v8h*)(Qr + kk * 32 + g * 8);
        v8h hi = *(const v8h*)(Qr + kk * 32 + 16 + g * 8);
#pragma unroll
        for (int i = 0; i < 8; ++i) { aQ[kk][i] = lo[i]; aQ[kk][8 + i] = hi[i]; }
    }

    v8f oacc[4] = {{}, {}, {}, {}};
    float mrow[8], lrow[8];
#pragma unroll
    for (int r = 0; r < 8; ++r) { mrow[r] = -3.0e38f; lrow[r] = 0.f; }

    const float scale = 0.03125f;  // C^-0.5 with C=1024 (per reference)
    const int send = qbase + 15;
    for (int s0 = 0; s0 <= send; s0 += 32) {
        // --- S = Q K^T for 32 keys (2 column tiles x 2 K-steps) ---
        v8f sc[2] = {{}, {}};
#pragma unroll
        for (int kk = 0; kk < 2; ++kk) {
#pragma unroll
            for (int nt = 0; nt < 2; ++nt) {
                const _Float16* Kr =
                    Kf + (bT + s0 + nt * 16 + lanen) * Cc + h64 + kk * 32 + g * 8;
                v8h klo = *(const v8h*)Kr;
                v8h khi = *(const v8h*)(Kr + 16);
                v16h bk;
#pragma unroll
                for (int i = 0; i < 8; ++i) { bk[i] = klo[i]; bk[8 + i] = khi[i]; }
                sc[nt] = WMMA_F32_F16(aQ[kk], bk, sc[nt]);
            }
        }
        // --- online softmax, write P tile (f16) to LDS ---
#pragma unroll
        for (int r = 0; r < 8; ++r) {
            const int row = qbase + r + 8 * g;
            const int c0 = s0 + lanen, c1 = c0 + 16;
            float v0 = (c0 <= row) ? sc[0][r] * scale : -3.0e38f;
            float v1 = (c1 <= row) ? sc[1][r] * scale : -3.0e38f;
            const float mt = redmax16(fmaxf(v0, v1));
            const float mn = fmaxf(mrow[r], mt);
            const float p0 = __expf(v0 - mn);
            const float p1 = __expf(v1 - mn);
            const float ps = redsum16(p0 + p1);
            const float alpha = __expf(mrow[r] - mn);
            lrow[r] = lrow[r] * alpha + ps;
            mrow[r] = mn;
#pragma unroll
            for (int dt = 0; dt < 4; ++dt) oacc[dt][r] *= alpha;
            P[(r + 8 * g) * 32 + lanen]      = (_Float16)p0;
            P[(r + 8 * g) * 32 + 16 + lanen] = (_Float16)p1;
        }
        asm volatile("s_wait_dscnt 0" ::: "memory");
        // --- reload P as A-fragment (C-layout -> A-layout transpose via LDS) ---
        v16h aP;
        {
            v8h lo = *(const v8h*)(P + lanen * 32 + g * 8);
            v8h hi = *(const v8h*)(P + lanen * 32 + 16 + g * 8);
#pragma unroll
            for (int i = 0; i < 8; ++i) { aP[i] = lo[i]; aP[8 + i] = hi[i]; }
        }
        // --- O += P V  (4 column tiles of DH=64) ---
#pragma unroll
        for (int dt = 0; dt < 4; ++dt) {
            v16h bv;
#pragma unroll
            for (int i = 0; i < 16; ++i)
                bv[i] = Vf[(bT + s0 + (i < 8 ? g * 8 + i : 16 + g * 8 + (i - 8))) * Cc + h64 + dt * 16 + lanen];
            oacc[dt] = WMMA_F32_F16(aP, bv, oacc[dt]);
        }
        asm volatile("s_wait_dscnt 0" ::: "memory");  // P reads done before overwrite
    }
#pragma unroll
    for (int r = 0; r < 8; ++r) lrow[r] = 1.0f / lrow[r];
#pragma unroll
    for (int dt = 0; dt < 4; ++dt) {
#pragma unroll
        for (int r = 0; r < 8; ++r) {
            const int row = qbase + r + 8 * g;
            Of[(bT + row) * Cc + h64 + dt * 16 + lanen] =
                (_Float16)(oacc[dt][r] * lrow[r]);
        }
    }
}

// ---------------------------------------------------------------------------
extern "C" void kernel_launch(void* const* d_in, const int* in_sizes, int n_in,
                              void* d_out, int out_size, void* d_ws, size_t ws_size,
                              hipStream_t stream) {
    (void)in_sizes; (void)n_in; (void)out_size; (void)ws_size;
    constexpr int Bb = 4, T = 1024, C = 1024, H = 16, DFF = 4096;
    constexpr int M = Bb * T;  // 4096

    const float* x  = (const float*)d_in[0];
    const float* g1 = (const float*)d_in[1];
    const float* g2 = (const float*)d_in[2];
    const float* wq = (const float*)d_in[3];
    const float* wk = (const float*)d_in[4];
    const float* wv = (const float*)d_in[5];
    const float* wp = (const float*)d_in[6];
    const float* bp = (const float*)d_in[7];
    const float* w1 = (const float*)d_in[8];
    const float* b1 = (const float*)d_in[9];
    const float* w2 = (const float*)d_in[10];
    const float* b2 = (const float*)d_in[11];

    char* ws = (char*)d_ws;
    size_t off = 0;
    auto alloc = [&](size_t bytes) -> void* {
        void* p = ws + off;
        off += (bytes + 255) & ~(size_t)255;
        return p;
    };
    _Float16* h1  = (_Float16*)alloc((size_t)M * C * 2);
    _Float16* qf  = (_Float16*)alloc((size_t)M * C * 2);
    _Float16* kf  = (_Float16*)alloc((size_t)M * C * 2);
    _Float16* vf  = (_Float16*)alloc((size_t)M * C * 2);
    _Float16* of  = (_Float16*)alloc((size_t)M * C * 2);
    float*    x1  = (float*)alloc((size_t)M * C * 4);
    _Float16* h2  = (_Float16*)alloc((size_t)M * C * 2);
    _Float16* f1  = (_Float16*)alloc((size_t)M * DFF * 2);
    _Float16* wqT = (_Float16*)alloc((size_t)C * C * 2);
    _Float16* wkT = (_Float16*)alloc((size_t)C * C * 2);
    _Float16* wvT = (_Float16*)alloc((size_t)C * C * 2);
    _Float16* wpT = (_Float16*)alloc((size_t)C * C * 2);
    _Float16* w1T = (_Float16*)alloc((size_t)C * DFF * 2);
    _Float16* w2T = (_Float16*)alloc((size_t)DFF * C * 2);

    const dim3 blk(256);
    // weight conversion (f32 row-major -> f16 col-major)
    cvt_transpose_f16<<<(C * C + 255) / 256, blk, 0, stream>>>(wq, wqT, C, C);
    cvt_transpose_f16<<<(C * C + 255) / 256, blk, 0, stream>>>(wk, wkT, C, C);
    cvt_transpose_f16<<<(C * C + 255) / 256, blk, 0, stream>>>(wv, wvT, C, C);
    cvt_transpose_f16<<<(C * C + 255) / 256, blk, 0, stream>>>(wp, wpT, C, C);
    cvt_transpose_f16<<<(C * DFF + 255) / 256, blk, 0, stream>>>(w1, w1T, C, DFF);
    cvt_transpose_f16<<<(DFF * C + 255) / 256, blk, 0, stream>>>(w2, w2T, DFF, C);

    // attention branch
    rmsnorm_f16<<<M, 256, 0, stream>>>(x, g1, h1);
    const dim3 gq(C / 128, M / 64);
    gemm_wmma<true, false, false, false><<<gq, blk, 0, stream>>>(h1, wqT, nullptr, nullptr, qf, M, C, C);
    gemm_wmma<true, false, false, false><<<gq, blk, 0, stream>>>(h1, wkT, nullptr, nullptr, kf, M, C, C);
    gemm_wmma<true, false, false, false><<<gq, blk, 0, stream>>>(h1, wvT, nullptr, nullptr, vf, M, C, C);
    attn_wmma<<<dim3(T / 128, H, Bb), blk, 0, stream>>>(qf, kf, vf, of);
    // x1 = x + (o @ w_proj + b_proj)
    gemm_wmma<false, false, true, true><<<gq, blk, 0, stream>>>(of, wpT, bp, x, x1, M, C, C);

    // FFN branch
    rmsnorm_f16<<<M, 256, 0, stream>>>(x1, g2, h2);
    gemm_wmma<true, true, true, false><<<dim3(DFF / 128, M / 64), blk, 0, stream>>>(h2, w1T, b1, nullptr, f1, M, DFF, C);
    // out = x1 + (f1 @ w2 + b2)
    gemm_wmma<false, false, true, true><<<gq, blk, 0, stream>>>(f1, w2T, b2, x1, d_out, M, C, DFF);
}
